// MultiHeadSelfAttention_635655160663
// MI455X (gfx1250) — matrix operand order repaired and checked by reading, not compiled, not run on hardware
//
#include <hip/hip_runtime.h>

// ---------------------------------------------------------------------------
// Multi-head self-attention for MI455X (gfx1250), wave32 + WMMA bf16.
//   q = x@Wq ; k = x@Wk ; v = x@Wv        (bf16 GEMM, f32 accum)
//   attn = softmax(q kT / sqrt(64)) v     (flash-style streaming, WMMA)
//   out  = attn@Wo                        (bf16 GEMM, f32 out)
// Staging uses GLOBAL_LOAD_ASYNC_TO_LDS_B128 (ASYNCcnt) when the builtin is
// available, falling back to VGPR-mediated copies otherwise.
// ---------------------------------------------------------------------------

typedef __attribute__((ext_vector_type(16))) __bf16 v16bf;
typedef __attribute__((ext_vector_type(8)))  __bf16 v8bf;
typedef __attribute__((ext_vector_type(8)))  float  v8f;

#define EMBED  1024
#define NHEAD  16
#define HDIM   64
#define BATCH  4
#define SEQ    2048
#define SCALE  0.125f   // 1/sqrt(HDIM)

#if defined(__has_builtin)
#if __has_builtin(__builtin_amdgcn_global_load_async_to_lds_b128)
#define HAVE_ASYNC_LDS 1
#endif
#endif
#ifndef HAVE_ASYNC_LDS
#define HAVE_ASYNC_LDS 0
#endif

#if HAVE_ASYNC_LDS
// Builtin prototype (from toolchain diagnostics): v4i AS1*, v4i AS3*, Ii, Ii
typedef int v4i_raw __attribute__((vector_size(16)));
typedef __attribute__((address_space(1))) v4i_raw* async_gptr_t;
typedef __attribute__((address_space(3))) v4i_raw* async_lptr_t;
// async copy of 16B per lane: LDS[lp+OFF] = MEM[gp+OFF]; tracked by ASYNCcnt
#define ASYNC_B128(gp, lp, OFF)                                              \
    __builtin_amdgcn_global_load_async_to_lds_b128(                          \
        (async_gptr_t)(gp), (async_lptr_t)(lp), (OFF), 0)
#define ASYNC_WAIT() asm volatile("s_wait_asynccnt 0x0" ::: "memory")
#endif

__device__ __forceinline__ unsigned short f32_to_bf16(float f) {
    union { float f; unsigned u; } v; v.f = f;
    unsigned u = v.u;
    u += 0x7FFFu + ((u >> 16) & 1u);          // round-to-nearest-even
    return (unsigned short)(u >> 16);
}

__device__ __forceinline__ v8f wmma_bf16f32(v16bf a, v16bf b, v8f c) {
    // (neg_a, A, neg_b, B, c_mod, C, reuse_a, reuse_b)
    return __builtin_amdgcn_wmma_f32_16x16x32_bf16(false, a, false, b,
                                                   (short)0, c, false, false);
}

// A-fragment (16x32 bf16, M x K), contiguous K order (matches B):
//   lanes 0-15 : row M=lane,    K = 0..15 in elems 0..15
//   lanes16-31 : row M=lane-16, K = 16..31
// rowPtr points at this lane's row, K index 0.  g = lane>>4.
__device__ __forceinline__ v16bf load_a_frag(const unsigned short* rowPtr, int g) {
    v8bf lo = *(const v8bf*)(rowPtr + g * 16);
    v8bf hi = *(const v8bf*)(rowPtr + g * 16 + 8);
    v16bf r;
#pragma unroll
    for (int i = 0; i < 8; ++i) { r[i] = lo[i]; r[8 + i] = hi[i]; }
    return r;
}

// B-fragment (32x16 bf16, K x N): lane n<16 holds col N=n, K=0..15;
// lanes 16-31 hold K=16..31.  colPtr points at BT[col][kbase], 16 contiguous.
__device__ __forceinline__ v16bf load_b_frag(const unsigned short* colPtr) {
    return *(const v16bf*)colPtr;
}

// ---------------------------------------------------------------------------
// fp32 -> bf16 cast
// ---------------------------------------------------------------------------
__global__ void cast_f32_to_bf16_kernel(const float* __restrict__ src,
                                        unsigned short* __restrict__ dst, int n) {
    int i = blockIdx.x * blockDim.x + threadIdx.x;
    if (i < n) dst[i] = f32_to_bf16(src[i]);
}

// fp32 [K][N] -> bf16 [N][K]  (K,N multiples of 16)
__global__ __launch_bounds__(256)
void transpose_f32_to_bf16_kernel(const float* __restrict__ W,
                                  unsigned short* __restrict__ WT, int K, int N) {
    __shared__ float tile[16][17];
    int n = blockIdx.x * 16 + threadIdx.x;
    int k = blockIdx.y * 16 + threadIdx.y;
    tile[threadIdx.y][threadIdx.x] = W[(size_t)k * N + n];
    __syncthreads();
    int on = blockIdx.x * 16 + threadIdx.y;
    int ok = blockIdx.y * 16 + threadIdx.x;
    WT[(size_t)on * K + ok] = f32_to_bf16(tile[threadIdx.x][threadIdx.y]);
}

// ---------------------------------------------------------------------------
// Tiled bf16 GEMM:  C[M][N] = A[M][K] * BT[N][K]^T + bias[N]
// Block: 256 threads = 8 waves; tile 128x128; each wave computes 32x64
// (2 M-frags x 4 N-frags of 16x16).  K-step 32 per LDS stage.
// ---------------------------------------------------------------------------
template <bool OUT_BF16>
__global__ __launch_bounds__(256)
void gemm_bf16_kernel(const unsigned short* __restrict__ A,
                      const unsigned short* __restrict__ BT,
                      const float* __restrict__ bias,
                      void* __restrict__ Cout,
                      int M, int N, int K) {
    __shared__ alignas(32) unsigned short As[128][32];
    __shared__ alignas(32) unsigned short Bs[128][32];

    const int t     = threadIdx.x;
    const int lane  = t & 31;
    const int wid   = t >> 5;         // 0..7
    const int waveM = wid >> 1;       // 0..3  -> 32 rows each
    const int waveN = wid & 1;        // 0..1  -> 64 cols each
    const int g     = lane >> 4;
    const int ln    = lane & 15;
    const int m0    = blockIdx.y * 128;
    const int n0    = blockIdx.x * 128;

    v8f acc[2][4];
#pragma unroll
    for (int mf = 0; mf < 2; ++mf)
#pragma unroll
        for (int nf = 0; nf < 4; ++nf)
#pragma unroll
            for (int i = 0; i < 8; ++i) acc[mf][nf][i] = 0.0f;

    const int row  = t >> 1;
    const int half = t & 1;
    const unsigned short* gArow = A  + (size_t)(m0 + row) * K + half * 16;
    const unsigned short* gBrow = BT + (size_t)(n0 + row) * K + half * 16;

    for (int kt = 0; kt < K; kt += 32) {
        // stage A tile (128x32) and BT tile (128x32): 32B per thread each
#if HAVE_ASYNC_LDS
        ASYNC_B128(gArow + kt, &As[row][half * 16], 0);
        ASYNC_B128(gArow + kt, &As[row][half * 16], 16);
        ASYNC_B128(gBrow + kt, &Bs[row][half * 16], 0);
        ASYNC_B128(gBrow + kt, &Bs[row][half * 16], 16);
#else
        {
            const uint4* src = (const uint4*)(gArow + kt);
            uint4* dst = (uint4*)(&As[row][half * 16]);
            dst[0] = src[0]; dst[1] = src[1];
        }
        {
            const uint4* src = (const uint4*)(gBrow + kt);
            uint4* dst = (uint4*)(&Bs[row][half * 16]);
            dst[0] = src[0]; dst[1] = src[1];
        }
#endif
        // speculative prefetch of next K-tile into the cache hierarchy
        if (kt + 32 < K) {
            __builtin_prefetch(gArow + kt + 32, 0, 1);
            __builtin_prefetch(gBrow + kt + 32, 0, 1);
        }
#if HAVE_ASYNC_LDS
        ASYNC_WAIT();              // drain this wave's ASYNCcnt before barrier
#endif
        __syncthreads();

        v16bf afrag[2], bfrag[4];
#pragma unroll
        for (int mf = 0; mf < 2; ++mf)
            afrag[mf] = load_a_frag(&As[waveM * 32 + mf * 16 + ln][0], g);
#pragma unroll
        for (int nf = 0; nf < 4; ++nf)
            bfrag[nf] = load_b_frag(&Bs[waveN * 64 + nf * 16 + ln][g * 16]);

#pragma unroll
        for (int mf = 0; mf < 2; ++mf)
#pragma unroll
            for (int nf = 0; nf < 4; ++nf)
                acc[mf][nf] = wmma_bf16f32(afrag[mf], bfrag[nf], acc[mf][nf]);

        __syncthreads();
    }

    // Epilogue: C-layout row = r + 8*g, col = ln within each 16x16 frag
#pragma unroll
    for (int mf = 0; mf < 2; ++mf)
#pragma unroll
        for (int nf = 0; nf < 4; ++nf)
#pragma unroll
            for (int r = 0; r < 8; ++r) {
                int rr = m0 + waveM * 32 + mf * 16 + r + 8 * g;
                int cc = n0 + waveN * 64 + nf * 16 + ln;
                float vv = acc[mf][nf][r] + bias[cc];
                if (OUT_BF16)
                    ((unsigned short*)Cout)[(size_t)rr * N + cc] = f32_to_bf16(vv);
                else
                    ((float*)Cout)[(size_t)rr * N + cc] = vv;
            }
}

// ---------------------------------------------------------------------------
// Flash-style attention.  Grid: (S/64, H, B), block 128 (4 waves).
// Each wave owns 16 q rows; workgroup streams 64-key blocks.
// Q/K/V stored bf16 [B*S][EMBED], head h at columns h*64..h*64+63.
// ---------------------------------------------------------------------------
__global__ __launch_bounds__(128)
void attention_kernel(const unsigned short* __restrict__ Q,
                      const unsigned short* __restrict__ Km,
                      const unsigned short* __restrict__ V,
                      unsigned short* __restrict__ Oout) {
    const int qt = blockIdx.x;
    const int h  = blockIdx.y;
    const int b  = blockIdx.z;
    const int t    = threadIdx.x;
    const int lane = t & 31;
    const int wid  = t >> 5;        // 0..3
    const int g    = lane >> 4;
    const int ln   = lane & 15;

    __shared__ alignas(32) unsigned short Qs[64][64];       // [q][d]
    __shared__ alignas(32) unsigned short Ks[64][64];       // [key][d]  (== BT for Q.K^T)
    __shared__ alignas(32) unsigned short Vt[64][64];       // [d][key]  (== BT for P.V)
    __shared__ alignas(32) unsigned short Ps[4][16][64];    // per-wave P [q][key]

    const int row  = t >> 1;
    const int half = t & 1;

    const unsigned short* gKrow = Km + ((size_t)(b * SEQ + row)) * EMBED + h * HDIM + half * 32;
    const unsigned short* gVrow = V  + ((size_t)(b * SEQ + row)) * EMBED + h * HDIM + half * 32;

    // stage Q tile (64 q x 64 d): 64B per thread
    {
        int s = qt * 64 + row;
        const unsigned short* gq = Q + ((size_t)(b * SEQ + s)) * EMBED + h * HDIM + half * 32;
#if HAVE_ASYNC_LDS
        ASYNC_B128(gq, &Qs[row][half * 32], 0);
        ASYNC_B128(gq, &Qs[row][half * 32], 16);
        ASYNC_B128(gq, &Qs[row][half * 32], 32);
        ASYNC_B128(gq, &Qs[row][half * 32], 48);
        ASYNC_WAIT();
#else
        const uint4* src = (const uint4*)gq;
        uint4* dst = (uint4*)(&Qs[row][half * 32]);
        dst[0] = src[0]; dst[1] = src[1]; dst[2] = src[2]; dst[3] = src[3];
#endif
    }
    __syncthreads();

    float mrow[8], lrow[8];
    v8f o[4];
#pragma unroll
    for (int r = 0; r < 8; ++r) { mrow[r] = -3.0e38f; lrow[r] = 0.0f; }
#pragma unroll
    for (int nf = 0; nf < 4; ++nf)
#pragma unroll
        for (int i = 0; i < 8; ++i) o[nf][i] = 0.0f;

    const int qbase = wid * 16;

    for (int kb0 = 0; kb0 < SEQ; kb0 += 64) {
        // stage K block (64 keys x 64 d), straight copy (64B per thread)
#if HAVE_ASYNC_LDS
        ASYNC_B128(gKrow + (size_t)kb0 * EMBED, &Ks[row][half * 32], 0);
        ASYNC_B128(gKrow + (size_t)kb0 * EMBED, &Ks[row][half * 32], 16);
        ASYNC_B128(gKrow + (size_t)kb0 * EMBED, &Ks[row][half * 32], 32);
        ASYNC_B128(gKrow + (size_t)kb0 * EMBED, &Ks[row][half * 32], 48);
#else
        {
            const uint4* src = (const uint4*)(gKrow + (size_t)kb0 * EMBED);
            uint4* dst = (uint4*)(&Ks[row][half * 32]);
            dst[0] = src[0]; dst[1] = src[1]; dst[2] = src[2]; dst[3] = src[3];
        }
#endif
        // stage V block transposed: Vt[d][key] (genuine transpose -> scalar)
        {
            const unsigned short* src = gVrow + (size_t)kb0 * EMBED;
#pragma unroll
            for (int i = 0; i < 32; ++i) Vt[half * 32 + i][row] = src[i];
        }
        // prefetch next key block while this one is consumed
        if (kb0 + 64 < SEQ) {
            __builtin_prefetch(gKrow + (size_t)(kb0 + 64) * EMBED, 0, 1);
            __builtin_prefetch(gVrow + (size_t)(kb0 + 64) * EMBED, 0, 1);
        }
#if HAVE_ASYNC_LDS
        ASYNC_WAIT();              // drain this wave's ASYNCcnt before barrier
#endif
        __syncthreads();

        // scores tile: 16 q x 64 keys, reduce over d=64 (2 WMMA K-steps)
        v8f sc[4];
#pragma unroll
        for (int nf = 0; nf < 4; ++nf)
#pragma unroll
            for (int i = 0; i < 8; ++i) sc[nf][i] = 0.0f;

#pragma unroll
        for (int ks = 0; ks < 2; ++ks) {
            v16bf af = load_a_frag(&Qs[qbase + ln][ks * 32], g);
#pragma unroll
            for (int nf = 0; nf < 4; ++nf) {
                v16bf bf_ = load_b_frag(&Ks[nf * 16 + ln][ks * 32 + g * 16]);
                sc[nf] = wmma_bf16f32(af, bf_, sc[nf]);
            }
        }

        // online softmax (rows striped: row = r + 8*g across 16-lane groups)
        float corr[8];
#pragma unroll
        for (int r = 0; r < 8; ++r) {
            float pm = fmaxf(fmaxf(sc[0][r], sc[1][r]), fmaxf(sc[2][r], sc[3][r])) * SCALE;
            pm = fmaxf(pm, __shfl_xor(pm, 1, 32));
            pm = fmaxf(pm, __shfl_xor(pm, 2, 32));
            pm = fmaxf(pm, __shfl_xor(pm, 4, 32));
            pm = fmaxf(pm, __shfl_xor(pm, 8, 32));
            float nm = fmaxf(mrow[r], pm);
            corr[r] = __expf(mrow[r] - nm);
            mrow[r] = nm;
        }

        float psum[8];
#pragma unroll
        for (int r = 0; r < 8; ++r) psum[r] = 0.0f;
#pragma unroll
        for (int nf = 0; nf < 4; ++nf)
#pragma unroll
            for (int r = 0; r < 8; ++r) {
                float p = __expf(sc[nf][r] * SCALE - mrow[r]);
                psum[r] += p;
                Ps[wid][r + 8 * g][nf * 16 + ln] = f32_to_bf16(p);
            }
#pragma unroll
        for (int r = 0; r < 8; ++r) {
            float s = psum[r];
            s += __shfl_xor(s, 1, 32);
            s += __shfl_xor(s, 2, 32);
            s += __shfl_xor(s, 4, 32);
            s += __shfl_xor(s, 8, 32);
            lrow[r] = lrow[r] * corr[r] + s;
#pragma unroll
            for (int nf = 0; nf < 4; ++nf) o[nf][r] *= corr[r];
        }
        __syncthreads();   // make P visible (uniform across waves)

        // O += P @ V : reduce over keys=64 (2 WMMA K-steps), N = d
#pragma unroll
        for (int ks = 0; ks < 2; ++ks) {
            v16bf pf = load_a_frag(&Ps[wid][ln][ks * 32], g);
#pragma unroll
            for (int nf = 0; nf < 4; ++nf) {
                v16bf vf = load_b_frag(&Vt[nf * 16 + ln][ks * 32 + g * 16]);
                o[nf] = wmma_bf16f32(pf, vf, o[nf]);
            }
        }
        __syncthreads();   // before restaging K/V
    }

    // finalize: divide by softmax denominator, write bf16 attn output
#pragma unroll
    for (int nf = 0; nf < 4; ++nf)
#pragma unroll
        for (int r = 0; r < 8; ++r) {
            int qrow = qt * 64 + qbase + r + 8 * g;
            int dcol = nf * 16 + ln;
            float val = o[nf][r] / lrow[r];
            Oout[((size_t)(b * SEQ + qrow)) * EMBED + h * HDIM + dcol] = f32_to_bf16(val);
        }
}

// ---------------------------------------------------------------------------
extern "C" void kernel_launch(void* const* d_in, const int* in_sizes, int n_in,
                              void* d_out, int out_size, void* d_ws, size_t ws_size,
                              hipStream_t stream) {
    (void)in_sizes; (void)n_in; (void)out_size; (void)ws_size;

    const float* x  = (const float*)d_in[0];
    const float* Wq = (const float*)d_in[1];
    const float* bq = (const float*)d_in[2];
    const float* Wk = (const float*)d_in[3];
    const float* bk = (const float*)d_in[4];
    const float* Wv = (const float*)d_in[5];
    const float* bv = (const float*)d_in[6];
    const float* Wo = (const float*)d_in[7];
    const float* bo = (const float*)d_in[8];
    float* out = (float*)d_out;

    const int M = BATCH * SEQ;           // 8192
    const int D = EMBED;                 // 1024

    // workspace carve-up (bf16 buffers)
    char* ws = (char*)d_ws;
    unsigned short* xb    = (unsigned short*)ws; ws += (size_t)M * D * 2;
    unsigned short* WqT   = (unsigned short*)ws; ws += (size_t)D * D * 2;
    unsigned short* WkT   = (unsigned short*)ws; ws += (size_t)D * D * 2;
    unsigned short* WvT   = (unsigned short*)ws; ws += (size_t)D * D * 2;
    unsigned short* WoT   = (unsigned short*)ws; ws += (size_t)D * D * 2;
    unsigned short* Qb    = (unsigned short*)ws; ws += (size_t)M * D * 2;
    unsigned short* Kb    = (unsigned short*)ws; ws += (size_t)M * D * 2;
    unsigned short* Vb    = (unsigned short*)ws; ws += (size_t)M * D * 2;
    unsigned short* attnb = (unsigned short*)ws; ws += (size_t)M * D * 2;

    // 1) casts / weight transposes
    {
        int n = M * D;
        cast_f32_to_bf16_kernel<<<(n + 255) / 256, 256, 0, stream>>>(x, xb, n);
        dim3 tb(16, 16), tg(D / 16, D / 16);
        transpose_f32_to_bf16_kernel<<<tg, tb, 0, stream>>>(Wq, WqT, D, D);
        transpose_f32_to_bf16_kernel<<<tg, tb, 0, stream>>>(Wk, WkT, D, D);
        transpose_f32_to_bf16_kernel<<<tg, tb, 0, stream>>>(Wv, WvT, D, D);
        transpose_f32_to_bf16_kernel<<<tg, tb, 0, stream>>>(Wo, WoT, D, D);
    }

    // 2) Q/K/V projections (bf16 out)
    {
        dim3 gg(D / 128, M / 128);   // (8, 64)
        gemm_bf16_kernel<true><<<gg, 256, 0, stream>>>(xb, WqT, bq, Qb, M, D, D);
        gemm_bf16_kernel<true><<<gg, 256, 0, stream>>>(xb, WkT, bk, Kb, M, D, D);
        gemm_bf16_kernel<true><<<gg, 256, 0, stream>>>(xb, WvT, bv, Vb, M, D, D);
    }

    // 3) attention
    {
        dim3 ag(SEQ / 64, NHEAD, BATCH);
        attention_kernel<<<ag, 128, 0, stream>>>(Qb, Kb, Vb, attnb);
    }

    // 4) output projection (fp32 out)
    {
        dim3 gg(D / 128, M / 128);
        gemm_bf16_kernel<false><<<gg, 256, 0, stream>>>(attnb, WoT, bo, out, M, D, D);
    }
}
